// Item_83760452206953
// MI455X (gfx1250) — matrix operand order repaired and checked by reading, not compiled, not run on hardware
//
#include <hip/hip_runtime.h>
#include <hip/hip_bf16.h>

typedef __attribute__((ext_vector_type(16))) _Float16     v16h;
typedef __attribute__((ext_vector_type(8)))  float        v8f;
typedef __attribute__((ext_vector_type(8)))  unsigned int v8u;

#define BROWS   2048
#define LF      64
#define NFIELD  5
#define OUTCOLS (NFIELD * LF)          // 320
#define CHUNK_K 2048                   // split-K chunk (multiple of 32)
#define WAVES_PER_WG 8
#define ROWS_PER_WG  (WAVES_PER_WG * 16)   // 128 rows per workgroup

// ---------------------------------------------------------------------------
// Kernel 0: zero the output (atomic split-K accumulates into it) and the
// per-field integer row-sum scratch at the head of d_ws.
// ---------------------------------------------------------------------------
__global__ void k_zero(float* __restrict__ out, int* __restrict__ sums) {
  int i = blockIdx.x * blockDim.x + threadIdx.x;
  if (i < BROWS * OUTCOLS) out[i] = 0.0f;
  if (i < NFIELD * BROWS)  sums[i] = 0;
}

// ---------------------------------------------------------------------------
// Kernel 1: convert f32 weights [K, 64] into f16 WMMA-B fragment order:
//   Wp[kt][n][kk] = (half) W[kt*32+k(kk)][n], k(kk) = kk with bits 3 and 4 swapped   (zero-padded past K)
// so a lane's 16-half B fragment (n fixed, kk = kh*16..kh*16+15) is one
// 32-byte contiguous run -> two global_load_b128, fully coalesced per wave.
// ---------------------------------------------------------------------------
__global__ void k_pack_weights(const float* __restrict__ W, _Float16* __restrict__ Wp,
                               int K, int Ktiles) {
  int i = blockIdx.x * blockDim.x + threadIdx.x;
  int total = Ktiles * LF * 32;
  if (i >= total) return;
  int kk = i & 31;
  int n  = (i >> 5) & (LF - 1);
  int kt = i >> 11;
  int k  = kt * 32 + ((kk & 7) | ((kk & 8) << 1) | ((kk & 16) >> 1));
  float v = (k < K) ? W[(size_t)k * LF + n] : 0.0f;
  Wp[i] = (_Float16)v;
}

// ---------------------------------------------------------------------------
// GEMM helpers
// ---------------------------------------------------------------------------
struct AInts { int4 a0, a1, a2, a3; };
struct BFrag { v16h b0, b1, b2, b3; };

// A-int slice for one K-step: halfs 0..7 <- A[m][k0+kh*8+e],
// halfs 8..15 <- A[m][k0+16+kh*8+e]  (abase already includes rowOff + kh*8)
__device__ __forceinline__ AInts loadA(const int* __restrict__ abase) {
  AInts r;
  r.a0 = reinterpret_cast<const int4*>(abase)[0];
  r.a1 = reinterpret_cast<const int4*>(abase)[1];
  r.a2 = reinterpret_cast<const int4*>(abase + 16)[0];
  r.a3 = reinterpret_cast<const int4*>(abase + 16)[1];
  return r;
}

__device__ __forceinline__ BFrag loadB(const _Float16* __restrict__ bt,
                                       int m, int kh) {
  BFrag r;
  r.b0 = *reinterpret_cast<const v16h*>(bt + ( 0 + m) * 32 + kh * 16);
  r.b1 = *reinterpret_cast<const v16h*>(bt + (16 + m) * 32 + kh * 16);
  r.b2 = *reinterpret_cast<const v16h*>(bt + (32 + m) * 32 + kh * 16);
  r.b3 = *reinterpret_cast<const v16h*>(bt + (48 + m) * 32 + kh * 16);
  return r;
}

// {0,1} ints -> packed f16 pair via (x | y<<16) * 0x3C00 (half(1.0)=0x3C00,
// no cross-half carry). The packed t also serves as a dual 16-bit row-sum
// counter (per-chunk counts <= 512 per half -> no overflow).
__device__ __forceinline__ v16h cvtA(const AInts& A, unsigned int& psum_pk) {
  v8u u;
  {
    unsigned t;
    t = (unsigned)A.a0.x | ((unsigned)A.a0.y << 16); psum_pk += t; u[0] = t * 0x3C00u;
    t = (unsigned)A.a0.z | ((unsigned)A.a0.w << 16); psum_pk += t; u[1] = t * 0x3C00u;
    t = (unsigned)A.a1.x | ((unsigned)A.a1.y << 16); psum_pk += t; u[2] = t * 0x3C00u;
    t = (unsigned)A.a1.z | ((unsigned)A.a1.w << 16); psum_pk += t; u[3] = t * 0x3C00u;
    t = (unsigned)A.a2.x | ((unsigned)A.a2.y << 16); psum_pk += t; u[4] = t * 0x3C00u;
    t = (unsigned)A.a2.z | ((unsigned)A.a2.w << 16); psum_pk += t; u[5] = t * 0x3C00u;
    t = (unsigned)A.a3.x | ((unsigned)A.a3.y << 16); psum_pk += t; u[6] = t * 0x3C00u;
    t = (unsigned)A.a3.z | ((unsigned)A.a3.w << 16); psum_pk += t; u[7] = t * 0x3C00u;
  }
  return __builtin_bit_cast(v16h, u);
}

// ---------------------------------------------------------------------------
// Kernel 2: tall-skinny GEMM via v_wmma_f32_16x16x32_f16, split-K,
// software-pipelined (iteration i+1 loads issued before iteration i compute).
// ---------------------------------------------------------------------------
__global__ __launch_bounds__(256)
void k_field_gemm(const int* __restrict__ mh, const _Float16* __restrict__ Wp,
                  float* __restrict__ out, int* __restrict__ sumsField,
                  int K, int K32, int colBase) {
  const int lane = threadIdx.x & 31;
  const int wave = threadIdx.x >> 5;
  const int m    = lane & 15;    // A row-in-tile / B,D column-in-tile
  const int kh   = lane >> 4;    // which K-half of the fragment this lane holds
  const int row0 = blockIdx.x * ROWS_PER_WG + wave * 16;
  const int row  = row0 + m;
  const size_t rowOff = (size_t)row * (size_t)K;

  const int kstart   = blockIdx.y * CHUNK_K;
  const int kend     = min(kstart + CHUNK_K, K32);
  const int kendFull = min(kend, K & ~31);   // last ragged tile only for K%32!=0

  v8f acc0 = {}, acc1 = {}, acc2 = {}, acc3 = {};
  unsigned int psum_pk = 0;
  int psum_tail = 0;

  // ---- branch-free pipelined main loop over full 32-wide K tiles ----
  if (kstart < kendFull) {
    const int* abase = mh + rowOff + kh * 8;
    AInts ca = loadA(abase + kstart);
    BFrag cb = loadB(Wp + (size_t)(kstart >> 5) * (LF * 32), m, kh);
    for (int k0 = kstart; k0 < kendFull; k0 += 32) {
      const int k1 = k0 + 32;
      const int kn = (k1 < kendFull) ? k1 : k0;   // clamped: last iter reloads k0
      AInts na = loadA(abase + kn);
      BFrag nb = loadB(Wp + (size_t)(kn >> 5) * (LF * 32), m, kh);
      __builtin_prefetch(abase + k0 + 512, 0, 1); // global_prefetch_b8, A stream

      v16h a = cvtA(ca, psum_pk);
      acc0 = __builtin_amdgcn_wmma_f32_16x16x32_f16(false, a, false, cb.b0, (short)0, acc0, false, false);
      acc1 = __builtin_amdgcn_wmma_f32_16x16x32_f16(false, a, false, cb.b1, (short)0, acc1, false, false);
      acc2 = __builtin_amdgcn_wmma_f32_16x16x32_f16(false, a, false, cb.b2, (short)0, acc2, false, false);
      acc3 = __builtin_amdgcn_wmma_f32_16x16x32_f16(false, a, false, cb.b3, (short)0, acc3, false, false);

      ca = na; cb = nb;
    }
  }

  // ---- ragged tail (only the decade field, K=12): guarded scalar A loads;
  // packed B is zero-padded so it needs no guards.
  if (kendFull < kend) {
    const int k0 = kendFull;
    v16h a;
    #pragma unroll
    for (int e = 0; e < 8; ++e) {
      int kA = k0 + kh * 8 + e;
      int kB = k0 + 16 + kh * 8 + e;
      int va = (kA < K) ? mh[rowOff + kA] : 0;
      int vb = (kB < K) ? mh[rowOff + kB] : 0;
      psum_tail += va + vb;
      a[e]     = (_Float16)va;
      a[8 + e] = (_Float16)vb;
    }
    BFrag cb = loadB(Wp + (size_t)(k0 >> 5) * (LF * 32), m, kh);
    acc0 = __builtin_amdgcn_wmma_f32_16x16x32_f16(false, a, false, cb.b0, (short)0, acc0, false, false);
    acc1 = __builtin_amdgcn_wmma_f32_16x16x32_f16(false, a, false, cb.b1, (short)0, acc1, false, false);
    acc2 = __builtin_amdgcn_wmma_f32_16x16x32_f16(false, a, false, cb.b2, (short)0, acc2, false, false);
    acc3 = __builtin_amdgcn_wmma_f32_16x16x32_f16(false, a, false, cb.b3, (short)0, acc3, false, false);
  }

  // ---- integer row sums: unpack dual counter, combine the two K-half lanes.
  int psum = (int)(psum_pk & 0xFFFFu) + (int)(psum_pk >> 16) + psum_tail;
  int other = __shfl_xor(psum, 16, 32);
  if (lane < 16) atomicAdd(&sumsField[row], psum + other);

  // ---- commit C tiles: D layout -> lane holds rows (r + 8*kh), column m.
  #pragma unroll
  for (int r = 0; r < 8; ++r) {
    float* o = out + (size_t)(row0 + r + 8 * kh) * OUTCOLS + colBase;
    atomicAdd(o + ( 0 + m), acc0[r]);
    atomicAdd(o + (16 + m), acc1[r]);
    atomicAdd(o + (32 + m), acc2[r]);
    atomicAdd(o + (48 + m), acc3[r]);
  }
}

// ---------------------------------------------------------------------------
// Kernel 3: normalization, faithfully replicating the reference's bug:
//  field 0 (decades): / sum_dec (if nz) then / sum_mov (if nz)
//  field 1 (movies):  untouched
//  fields 2..4:       / own sum (if nz)
// ---------------------------------------------------------------------------
__global__ void k_normalize(float* __restrict__ out, const int* __restrict__ sums) {
  int i = blockIdx.x * blockDim.x + threadIdx.x;
  if (i >= BROWS * OUTCOLS) return;
  int row = i / OUTCOLS;
  int col = i - row * OUTCOLS;
  int f   = col >> 6;
  float v = out[i];
  if (f == 0) {
    int sd = sums[row];
    int sm = sums[BROWS + row];
    if (sd != 0) v /= (float)sd;
    if (sm != 0) v /= (float)sm;
  } else if (f >= 2) {
    int s = sums[f * BROWS + row];
    if (s != 0) v /= (float)s;
  }
  out[i] = v;
}

// ---------------------------------------------------------------------------
extern "C" void kernel_launch(void* const* d_in, const int* in_sizes, int n_in,
                              void* d_out, int out_size, void* d_ws, size_t ws_size,
                              hipStream_t stream) {
  (void)in_sizes; (void)n_in; (void)out_size; (void)ws_size;

  const int*   mhs[NFIELD] = { (const int*)d_in[0], (const int*)d_in[1],
                               (const int*)d_in[2], (const int*)d_in[3],
                               (const int*)d_in[4] };
  const float* Ws[NFIELD]  = { (const float*)d_in[5], (const float*)d_in[6],
                               (const float*)d_in[7], (const float*)d_in[8],
                               (const float*)d_in[9] };
  const int Ks[NFIELD] = { 12, 60000, 32, 100000, 20000 };

  float* out  = (float*)d_out;
  int*   sums = (int*)d_ws;                               // [5][2048] ints
  _Float16* wpackBase =
      (_Float16*)((char*)d_ws + (size_t)NFIELD * BROWS * sizeof(int));

  // Zero output + sums.
  {
    int total = BROWS * OUTCOLS;                          // covers sums region too
    k_zero<<<(total + 255) / 256, 256, 0, stream>>>(out, sums);
  }

  // Pack weights (f32 -> f16 fragment order) into workspace.
  size_t halfOff[NFIELD];
  int    tiles[NFIELD];
  {
    size_t off = 0;
    for (int f = 0; f < NFIELD; ++f) {
      tiles[f]   = (Ks[f] + 31) / 32;
      halfOff[f] = off;
      off += (size_t)tiles[f] * LF * 32;
    }
    for (int f = 0; f < NFIELD; ++f) {
      int total = tiles[f] * LF * 32;
      k_pack_weights<<<(total + 255) / 256, 256, 0, stream>>>(
          Ws[f], wpackBase + halfOff[f], Ks[f], tiles[f]);
    }
  }

  // Split-K WMMA GEMMs, one launch per field (stream-ordered after packing).
  for (int f = 0; f < NFIELD; ++f) {
    int K32     = tiles[f] * 32;
    int nchunks = (K32 + CHUNK_K - 1) / CHUNK_K;
    dim3 grid(BROWS / ROWS_PER_WG, nchunks);
    k_field_gemm<<<grid, 256, 0, stream>>>(
        mhs[f], wpackBase + halfOff[f], out, sums + (size_t)f * BROWS,
        Ks[f], K32, f * LF);
  }

  // Normalization epilogue.
  k_normalize<<<(BROWS * OUTCOLS + 255) / 256, 256, 0, stream>>>(out, sums);
}
